// EmbeddingBlock_18562848654096
// MI455X (gfx1250) — matrix operand order repaired and checked by reading, not compiled, not run on hardware
//
#include <hip/hip_runtime.h>
#include <hip/hip_bf16.h>
#include <stdint.h>

typedef __attribute__((ext_vector_type(16))) __bf16 v16bf;
typedef __attribute__((ext_vector_type(8)))  float  v8f;

#define H       128
#define E_EDGES 1000000
#define NTILES  (E_EDGES / 16)   // 62500 exactly

static __device__ __forceinline__ uint16_t f2bf_bits(float f) {
  union { __hip_bfloat16 h; uint16_t u; } cv;
  cv.h = __float2bfloat16(f);
  return cv.u;
}
// packed pair: low half = a, high half = b
static __device__ __forceinline__ uint32_t pk2bf(float a, float b) {
  return (uint32_t)f2bf_bits(a) | ((uint32_t)f2bf_bits(b) << 16);
}
// fast silu: v * v_rcp_f32(1+exp(-v)) — avoids IEEE div expansion
static __device__ __forceinline__ float silu(float v) {
  return v * __builtin_amdgcn_rcpf(1.0f + __expf(-v));
}

// ---------------------------------------------------------------------------
// Precompute 1: T1b[v][n] = emb_table[v]·W_emb[0:128][n] + b_emb[n]
//               T2 [v][n] = emb_table[v]·W_emb[128:256][n]
// ---------------------------------------------------------------------------
__global__ void precompute_tables(const float* __restrict__ emb_table,
                                  const float* __restrict__ W_emb,
                                  const float* __restrict__ b_emb,
                                  float* __restrict__ T1b,
                                  float* __restrict__ T2) {
  const int v = blockIdx.x;     // vocab entry
  const int n = threadIdx.x;    // output column
  float a1 = b_emb[n];
  float a2 = 0.0f;
  for (int k = 0; k < H; ++k) {
    const float e = emb_table[v * H + k];
    a1 += e * W_emb[k * H + n];
    a2 += e * W_emb[(H + k) * H + n];
  }
  T1b[v * H + n] = a1;
  T2[v * H + n]  = a2;
}

// ---------------------------------------------------------------------------
// Precompute 2: per-lane register images of the B operands (bf16), matching
// a contiguous-k 32x16 B order (same k order as the GEMM2 A reload):
//   VGPR v, lane n<16 : B[K=2v,   nt*16+n], B[K=2v+1, ...]
//   VGPR v, lane n>=16: B[K=16+2v, ...  ],  B[K=17+2v, ...]
// W3img: 32 (c,nt) combos, K' = 32c + layout-K  (W3 = W_emb rows 256..383)
// W1img: 8 nt combos, K padded 16->32 (upper half zero) for W_pair
// ---------------------------------------------------------------------------
__global__ void precompute_img(const float* __restrict__ W_pair,
                               const float* __restrict__ W_emb,
                               uint32_t* __restrict__ W3img,
                               uint32_t* __restrict__ W1img) {
  const int t = blockIdx.x * blockDim.x + threadIdx.x;
  if (t < 8192) {                       // W3: [cmb=c*8+nt][lane][v] dwords
    const int cmb = t >> 8, rem = t & 255, lane = rem >> 3, v = rem & 7;
    const int c = cmb >> 3, nt = cmb & 7;
    const int k0 = 32 * c + 2 * v + ((lane >= 16) ? 16 : 0);
    const int n  = nt * 16 + (lane & 15);
    const float a = W_emb[(256 + k0) * H + n];
    const float b = W_emb[(256 + k0 + 1) * H + n];
    W3img[t] = pk2bf(a, b);
  } else if (t < 8192 + 2048) {         // W_pair padded: [nt][lane][v]
    const int u = t - 8192;
    const int nt = u >> 8, rem = u & 255, lane = rem >> 3, v = rem & 7;
    uint32_t val = 0;
    if (v < 4) {                        // K = 8*(lane>=16) + 2v, +1 in [0,16); v>=4 is zero pad
      const int n = nt * 16 + (lane & 15);
      const int k = 2 * v + ((lane >= 16) ? 8 : 0);
      val = pk2bf(W_pair[k * H + n], W_pair[(k + 1) * H + n]);
    }
    W1img[u] = val;
  }
}

// ---------------------------------------------------------------------------
// Main edge kernel. One wave = one 16-edge tile per loop iteration.
//   GEMM1: pbh = silu(pair_basis·W_pair + b_pair)   (8x  wmma bf16, K padded)
//   GEMM2: acc = pbh·W3                             (32x wmma bf16, fp32 acc)
//   out   = silu(acc + T1b[x[i]] + T2[x[j]])
// ---------------------------------------------------------------------------
__global__ void __launch_bounds__(256)
edge_mlp(const int*   __restrict__ gx,
         const float* __restrict__ pair_basis,
         const int*   __restrict__ gi,
         const int*   __restrict__ gj,
         const float* __restrict__ b_pair,
         const float* __restrict__ T1b,
         const float* __restrict__ T2,
         const uint32_t* __restrict__ W3img,
         const uint32_t* __restrict__ W1img,
         float* __restrict__ out) {
  const int lane  = threadIdx.x & 31;
  const int wave  = threadIdx.x >> 5;
  const int hi    = lane >> 4;        // 0: lanes 0-15, 1: lanes 16-31
  const int l15   = lane & 15;
  const int gwave = blockIdx.x * 8 + wave;
  const int nwv   = gridDim.x * 8;

  // Per-wave private pbh staging slice; row stride 136 (=272B -> 4-dword bank
  // skew) so the 16-row strided A-layout reloads are conflict free.
  __shared__ __align__(16) uint16_t pbh[8][16][136];

  // ---- one-time per-wave operand preload (stays resident in VGPRs) ----
  v16bf B3[4][8];
#pragma unroll
  for (int c = 0; c < 4; ++c)
#pragma unroll
    for (int nt = 0; nt < 8; ++nt)
      B3[c][nt] = *reinterpret_cast<const v16bf*>(W3img + (((c * 8 + nt) * 32 + lane) * 8));

  v16bf B1[8];
#pragma unroll
  for (int nt = 0; nt < 8; ++nt)
    B1[nt] = *reinterpret_cast<const v16bf*>(W1img + ((nt * 32 + lane) * 8));

  float bp[8];
#pragma unroll
  for (int nt = 0; nt < 8; ++nt) bp[nt] = b_pair[nt * 16 + l15];

  const v8f vzero = {};

  for (int t = gwave; t < NTILES; t += nwv) {
    const int row0  = t * 16;
    const int myrow = row0 + l15;

    if (t + nwv < NTILES)  // global_prefetch_b8 next tile's pair_basis (1KB)
      __builtin_prefetch(pair_basis + (size_t)(t + nwv) * 256 + lane * 8, 0, 0);

    // ---- GEMM1: A = pair_basis tile in 16-bit A layout, K padded to 32 ----
    // lane<16: V0-3 = pb[row,0..7], V4-7 = 0 ; lane>=16: V0-3 = pb[row,8..15]
    const float* pbp = pair_basis + (size_t)myrow * 16 + hi * 8;
    const float4 f0 = *reinterpret_cast<const float4*>(pbp);
    const float4 f1 = *reinterpret_cast<const float4*>(pbp + 4);
    union { v16bf v; uint32_t d[8]; } au;
    au.d[0] = pk2bf(f0.x, f0.y);
    au.d[1] = pk2bf(f0.z, f0.w);
    au.d[2] = pk2bf(f1.x, f1.y);
    au.d[3] = pk2bf(f1.z, f1.w);
    au.d[4] = 0; au.d[5] = 0; au.d[6] = 0; au.d[7] = 0;

#pragma unroll
    for (int nt = 0; nt < 8; ++nt) {
      v8f d = __builtin_amdgcn_wmma_f32_16x16x32_bf16(
          false, au.v, false, B1[nt], (short)0, vzero, false, false);
#pragma unroll
      for (int r = 0; r < 8; ++r) {           // D: VGPR r = row r / r+8
        float x = d[r] + bp[nt];
        x = silu(x);
        pbh[wave][r + hi * 8][nt * 16 + l15] = f2bf_bits(x);
      }
    }

    // ---- GEMM2: reload pbh in A layout, 4 K-chunks x 8 N-tiles ----
    v8f acc[8];
#pragma unroll
    for (int nt = 0; nt < 8; ++nt) acc[nt] = vzero;

#pragma unroll
    for (int c = 0; c < 4; ++c) {
      const uint16_t* ap = &pbh[wave][l15][c * 32 + hi * 16];
      union { v16bf v; uint4 q[2]; } a2;
      a2.q[0] = *reinterpret_cast<const uint4*>(ap);        // K' 0..7 / 16..23
      a2.q[1] = *reinterpret_cast<const uint4*>(ap + 8);    // K' 8..15 / 24..31
#pragma unroll
      for (int nt = 0; nt < 8; ++nt)
        acc[nt] = __builtin_amdgcn_wmma_f32_16x16x32_bf16(
            false, a2.v, false, B3[c][nt], (short)0, acc[nt], false, false);
    }

    // ---- epilogue: + T1b[x[i]] + T2[x[j]], silu, store ----
    // lanes 0-15 fetch x[i[row]] for rows 0..15; lanes 16-31 fetch x[j[row]].
    const int* idxp = (lane < 16) ? (gi + row0 + lane) : (gj + row0 + l15);
    const int  xv   = gx[*idxp];
#pragma unroll
    for (int r = 0; r < 8; ++r) {
      const int ri = r + hi * 8;                 // tile row held by VGPR r
      const int xi = __shfl(xv, ri, 32);
      const int xj = __shfl(xv, ri + 16, 32);
      const float* t1p = T1b + (size_t)xi * H + l15;
      const float* t2p = T2  + (size_t)xj * H + l15;
      float*       op  = out + (size_t)(row0 + ri) * H + l15;
#pragma unroll
      for (int nt = 0; nt < 8; ++nt) {
        float x = acc[nt][r] + t1p[nt * 16] + t2p[nt * 16];
        op[nt * 16] = silu(x);
      }
    }
  }
}

// ---------------------------------------------------------------------------
extern "C" void kernel_launch(void* const* d_in, const int* in_sizes, int n_in,
                              void* d_out, int out_size, void* d_ws, size_t ws_size,
                              hipStream_t stream) {
  const int*   x          = (const int*)  d_in[0];
  const float* pair_basis = (const float*)d_in[1];
  const int*   i_         = (const int*)  d_in[2];
  const int*   j_         = (const int*)  d_in[3];
  const float* emb_table  = (const float*)d_in[4];
  const float* W_pair     = (const float*)d_in[5];
  const float* b_pair     = (const float*)d_in[6];
  const float* W_emb      = (const float*)d_in[7];
  const float* b_emb      = (const float*)d_in[8];
  float* out = (float*)d_out;

  // workspace layout (148,480 bytes)
  float*    ws    = (float*)d_ws;
  float*    T1b   = ws;                        // 105*128 f32
  float*    T2    = ws + 105 * H;              // 105*128 f32
  uint32_t* W3img = (uint32_t*)(ws + 2 * 105 * H);  // 8192 dwords
  uint32_t* W1img = W3img + 8192;                    // 2048 dwords

  precompute_tables<<<105, H, 0, stream>>>(emb_table, W_emb, b_emb, T1b, T2);
  precompute_img<<<40, 256, 0, stream>>>(W_pair, W_emb, W3img, W1img);
  edge_mlp<<<512, 256, 0, stream>>>(x, pair_basis, i_, j_, b_pair,
                                    T1b, T2, W3img, W1img, out);
}
